// MultiheadSelfAttention_7121055777554
// MI455X (gfx1250) — matrix operand order repaired and checked by reading, not compiled, not run on hardware
//
#include <hip/hip_runtime.h>
#include <hip/hip_bf16.h>

// ---------------------------------------------------------------------------
// MultiheadSelfAttention for MI455X (gfx1250): bf16 WMMA everywhere,
// register double-buffered K-loops so v_wmma does not stall on loadcnt 0.
// B=2, S=2048, D=1024, H=16, DK=64.
// ---------------------------------------------------------------------------

constexpr int B  = 2;
constexpr int S  = 2048;
constexpr int D  = 1024;
constexpr int H  = 16;
constexpr int DK = 64;

typedef __attribute__((ext_vector_type(16))) __bf16 v16bf;
typedef __attribute__((ext_vector_type(8)))  __bf16 v8bf;
typedef __attribute__((ext_vector_type(8)))  float  v8f;
typedef __attribute__((ext_vector_type(4)))  float  v4f;

// ---- bf16 helpers (bit-level, no libcalls) --------------------------------
__device__ __forceinline__ unsigned short f2bf_bits(float f) {
  unsigned u = __builtin_bit_cast(unsigned, f);
  u += 0x7FFFu + ((u >> 16) & 1u);          // round-to-nearest-even
  return (unsigned short)(u >> 16);
}
__device__ __forceinline__ __bf16 f2bf(float f) {
  unsigned short h = f2bf_bits(f);
  return __builtin_bit_cast(__bf16, h);
}

// ---- WMMA fragment loaders (ISA 7.12.2 layouts, wave32) -------------------
// A-matrix 16x32 bf16: lane<16 holds row r=lane, K = k0+0..7 and k0+16..23;
// lane>=16 holds row r=lane-16, K = k0+8..15 and k0+24..31.
__device__ __forceinline__ v16bf load_a_frag(const __bf16* base, int ld,
                                             int row0, int k0) {
  const int lane = threadIdx.x & 31;
  const int r    = lane & 15;
  const int half = lane >> 4;
  const __bf16* p = base + (size_t)(row0 + r) * ld + k0 + half * 8;
  v8bf lo = *(const v8bf*)(p);
  v8bf hi = *(const v8bf*)(p + 16);
  v16bf f;
#pragma unroll
  for (int i = 0; i < 8; ++i) { f[i] = lo[i]; f[i + 8] = hi[i]; }
  return f;
}

// B-matrix 32x16 bf16 supplied as N x K rows ("Bt"): lane holds column
// N = lane&15, K = k0+0..7,16..23 (lane<16) or k0+8..15,24..31 (lane>=16).
__device__ __forceinline__ v16bf load_b_frag(const __bf16* base, int ld,
                                             int n0, int k0) {
  const int lane = threadIdx.x & 31;
  const int n    = lane & 15;
  const int half = lane >> 4;
  const __bf16* p = base + (size_t)(n0 + n) * ld + k0 + half * 8;
  v8bf lo = *(const v8bf*)(p);
  v8bf hi = *(const v8bf*)(p + 16);
  v16bf f;
#pragma unroll
  for (int i = 0; i < 8; ++i) { f[i] = lo[i]; f[i + 8] = hi[i]; }
  return f;
}

__device__ __forceinline__ v8f wmma_bf16(v16bf a, v16bf b, v8f c) {
  return __builtin_amdgcn_wmma_f32_16x16x32_bf16(
      /*neg_a=*/false, a, /*neg_b=*/false, b,
      /*c_mod=*/(short)0, c, /*reuse_a=*/false, /*reuse_b=*/false);
}

// ---------------------------------------------------------------------------
// Kernel 1: f32 -> bf16 conversion, 4 elements / thread
// ---------------------------------------------------------------------------
__global__ void cvt_f32_bf16_kernel(const float* __restrict__ src,
                                    __bf16* __restrict__ dst, int n4) {
  int i = blockIdx.x * blockDim.x + threadIdx.x;
  if (i < n4) {
    v4f v = *(const v4f*)(src + 4 * (size_t)i);
    unsigned short h[4];
#pragma unroll
    for (int j = 0; j < 4; ++j) h[j] = f2bf_bits(v[j]);
    unsigned lo = (unsigned)h[0] | ((unsigned)h[1] << 16);
    unsigned hi = (unsigned)h[2] | ((unsigned)h[3] << 16);
    unsigned long long pk = (unsigned long long)lo | ((unsigned long long)hi << 32);
    *(unsigned long long*)(dst + 4 * (size_t)i) = pk;
  }
}

// ---------------------------------------------------------------------------
// Kernel 2: fused Q/K/V projection + RoPE.  One wave = 16x16 tile x {Q,K,V}.
//   Q,K -> [B,H,S,DK] bf16 (RoPE applied);  V -> [B,H,DK,S] bf16 (transposed)
//   K-loop is register double-buffered: loads for step k+32 are issued
//   before the WMMAs of step k.
// ---------------------------------------------------------------------------
__global__ void __launch_bounds__(32)
proj_qkv_kernel(const __bf16* __restrict__ xb, const __bf16* __restrict__ wq,
                const __bf16* __restrict__ wk, const __bf16* __restrict__ wv,
                const int* __restrict__ pos,
                __bf16* __restrict__ Qh, __bf16* __restrict__ Kh,
                __bf16* __restrict__ Vt) {
  __shared__ float tile[16][17];

  const int n0   = blockIdx.x * 16;   // output feature column (o)
  const int r0   = blockIdx.y * 16;   // flattened (b,s) row
  const int lane = threadIdx.x & 31;

  v8f cq = {}, ck = {}, cv = {};

  // prologue
  v16bf a  = load_a_frag(xb, D, r0, 0);
  v16bf bq = load_b_frag(wq, D, n0, 0);
  v16bf bk = load_b_frag(wk, D, n0, 0);
  v16bf bv = load_b_frag(wv, D, n0, 0);

  for (int k0 = 0; k0 < D - 32; k0 += 32) {
    // issue next-step loads first (overlaps with the WMMAs below)
    v16bf a2  = load_a_frag(xb, D, r0, k0 + 32);
    v16bf bq2 = load_b_frag(wq, D, n0, k0 + 32);
    v16bf bk2 = load_b_frag(wk, D, n0, k0 + 32);
    v16bf bv2 = load_b_frag(wv, D, n0, k0 + 32);
    cq = wmma_bf16(a, bq, cq);
    ck = wmma_bf16(a, bk, ck);
    cv = wmma_bf16(a, bv, cv);
    a = a2; bq = bq2; bk = bk2; bv = bv2;
  }
  cq = wmma_bf16(a, bq, cq);
  ck = wmma_bf16(a, bk, ck);
  cv = wmma_bf16(a, bv, cv);

  const int b      = r0 / S;
  const int s_base = r0 % S;
  const int h      = n0 / DK;
  const int dk0    = n0 % DK;   // 16-col tile lies within one head

  const int nn   = lane & 15;
  const int mrow = (lane >> 4) * 8;

  // dump C fragment to LDS: lane = N, VGPR r = M (=r or r+8)
  auto dump = [&](const v8f& c) {
#pragma unroll
    for (int r = 0; r < 8; ++r) tile[mrow + r][nn] = c[r];
  };

  // RoPE + packed bf16x2 store to [B,H,S,DK]
  auto rope_store = [&](__bf16* out) {
#pragma unroll
    for (int it = 0; it < 4; ++it) {
      int p = lane + 32 * it;        // 128 (row,pair) slots
      int m = p >> 3;
      int j = p & 7;                 // cols 2j, 2j+1
      int s = s_base + m;
      int i = (dk0 + 2 * j) >> 1;
      float fr  = exp2f(-((float)(2 * i) / 64.0f) * 13.287712379549449f);
      float ang = (float)pos[s] * fr;
      float sn, cs;
      __sincosf(ang, &sn, &cs);
      float x1 = tile[m][2 * j];
      float x2 = tile[m][2 * j + 1];
      float r1 = x1 * cs - x2 * sn;
      float r2 = x1 * sn + x2 * cs;
      unsigned pk = (unsigned)f2bf_bits(r1) | ((unsigned)f2bf_bits(r2) << 16);
      size_t idx = (((size_t)(b * H + h) * S + s) * DK + dk0 + 2 * j);
      *(unsigned*)(out + idx) = pk;
    }
  };

  dump(cq); __syncthreads(); rope_store(Qh); __syncthreads();
  dump(ck); __syncthreads(); rope_store(Kh); __syncthreads();
  dump(cv); __syncthreads();
  // V: transposed store -> Vt[((b*H+h)*DK + dk)*S + s]
#pragma unroll
  for (int it = 0; it < 8; ++it) {
    int e = lane + 32 * it;          // 256 elements
    int m = e & 15;                  // s (fast, coalesced)
    int n = e >> 4;                  // dk
    size_t idx = ((size_t)(b * H + h) * DK + dk0 + n) * S + (s_base + m);
    Vt[idx] = f2bf(tile[m][n]);
  }
}

// ---------------------------------------------------------------------------
// Kernel 3: causal flash attention.  One wave = 16 queries of one (b,h).
//   S^T = K·Q^T  (lane = query -> cheap softmax; P^T feeds PV A-frag with
//   zero cross-lane traffic).  K frags double-buffered across key tiles;
//   V frags issued before the softmax VALU block to hide their latency.
// ---------------------------------------------------------------------------
__global__ void __launch_bounds__(32)
flash_attn_kernel(const __bf16* __restrict__ Qh, const __bf16* __restrict__ Kh,
                  const __bf16* __restrict__ Vt, __bf16* __restrict__ Oc) {
  constexpr int QT = S / 16;         // query tiles per sequence
  const int wid = blockIdx.x;
  const int qt  = wid % QT;
  const int h   = (wid / QT) % H;
  const int b   = wid / (QT * H);
  const int q0  = qt * 16;

  const int lane   = threadIdx.x & 31;
  const int qn     = lane & 15;      // this lane's query (softmax role)
  const int q_glob = q0 + qn;
  const int mrow   = (lane >> 4) * 8;

  const __bf16* Qb = Qh + (size_t)(b * H + h) * S * DK;
  const __bf16* Kb = Kh + (size_t)(b * H + h) * S * DK;
  const __bf16* Vb = Vt + (size_t)(b * H + h) * DK * S;

  // Q as B-fragments (N = query, K = dk), loaded once
  v16bf qf0 = load_b_frag(Qb, DK, q0, 0);
  v16bf qf1 = load_b_frag(Qb, DK, q0, 32);

  v8f o0 = {}, o1 = {}, o2 = {}, o3 = {};
  float m_run = -1e30f, l_run = 0.0f;
  const float sscale = 0.125f;       // 1/sqrt(64)
  const int kmax = q0 + 15;

  // prologue: K fragments of first key tile
  v16bf ka0 = load_a_frag(Kb, DK, 0, 0);
  v16bf ka1 = load_a_frag(Kb, DK, 0, 32);
  v16bf kb0 = load_a_frag(Kb, DK, 16, 0);
  v16bf kb1 = load_a_frag(Kb, DK, 16, 32);

  for (int k0 = 0; k0 <= kmax; k0 += 32) {
    // scores^T: keys k0..k0+15 (sa) and k0+16..k0+31 (sb)
    v8f sa = {};
    sa = wmma_bf16(ka0, qf0, sa);
    sa = wmma_bf16(ka1, qf1, sa);
    v8f sb = {};
    sb = wmma_bf16(kb0, qf0, sb);
    sb = wmma_bf16(kb1, qf1, sb);

    // issue next-tile K loads + this tile's V loads before the softmax VALU
    // (row index clamped so the final-iteration load stays in bounds; its
    //  values are never consumed)
    const int kn = (k0 + 32 <= S - 32) ? (k0 + 32) : 0;
    __builtin_prefetch(Kb + (size_t)(kn)*DK, 0, 0);
    v16bf na0 = load_a_frag(Kb, DK, kn, 0);
    v16bf na1 = load_a_frag(Kb, DK, kn, 32);
    v16bf nb0 = load_a_frag(Kb, DK, kn + 16, 0);
    v16bf nb1 = load_a_frag(Kb, DK, kn + 16, 32);
    v16bf v0 = load_b_frag(Vb, S, 0,  k0);
    v16bf v1 = load_b_frag(Vb, S, 16, k0);
    v16bf v2 = load_b_frag(Vb, S, 32, k0);
    v16bf v3 = load_b_frag(Vb, S, 48, k0);

    // causal mask + scale + running max (one xor-16 combine: lane pair covers
    // all 32 keys of this lane's query)
    float sav[8], sbv[8];
    float tmax = -1e30f;
#pragma unroll
    for (int r = 0; r < 8; ++r) {
      int ka = k0 + mrow + r;
      int kb = k0 + 16 + mrow + r;
      float va = (ka <= q_glob) ? sa[r] * sscale : -1e30f;
      float vb = (kb <= q_glob) ? sb[r] * sscale : -1e30f;
      sav[r] = va; sbv[r] = vb;
      tmax = fmaxf(tmax, fmaxf(va, vb));
    }
    tmax = fmaxf(tmax, __shfl_xor(tmax, 16));
    float m_new = fmaxf(m_run, tmax);
    float scale = __expf(m_run - m_new);

    // exp weights -> P^T C-layout maps directly into the PV A-fragment:
    //   lane<16 needs K = 0..7,16..23  = local sa,sb VGPRs
    //   lane>=16 needs K = 8..15,24..31 = local sa,sb VGPRs   (no shuffles!)
    float tsum = 0.0f;
    v16bf pf;
#pragma unroll
    for (int r = 0; r < 8; ++r) {
      float pa = __expf(sav[r] - m_new);
      float pb = __expf(sbv[r] - m_new);
      tsum += pa + pb;
      pf[r]     = f2bf(pa);
      pf[r + 8] = f2bf(pb);
    }
    tsum += __shfl_xor(tsum, 16);
    l_run = l_run * scale + tsum;
    m_run = m_new;

    // rescale O accumulators: per-query scale broadcast into C-layout
#pragma unroll
    for (int r = 0; r < 8; ++r) {
      float sc = __shfl(scale, mrow + r);  // query m=mrow+r lives in lane m
      o0[r] *= sc; o1[r] *= sc; o2[r] *= sc; o3[r] *= sc;
    }

    // P·V (V frags issued above, latency hidden by the softmax math)
    o0 = wmma_bf16(pf, v0, o0);
    o1 = wmma_bf16(pf, v1, o1);
    o2 = wmma_bf16(pf, v2, o2);
    o3 = wmma_bf16(pf, v3, o3);

    ka0 = na0; ka1 = na1; kb0 = nb0; kb1 = nb1;
  }

  // finalize: divide by l, store bf16 to Oc[b, s, h*64 + dk]
  float linv = 1.0f / l_run;
#pragma unroll
  for (int r = 0; r < 8; ++r) {
    float li = __shfl(linv, mrow + r);
    int s = q0 + mrow + r;
    size_t base = ((size_t)b * S + s) * D + h * DK;
    int n = lane & 15;
    Oc[base + 0  + n] = f2bf(o0[r] * li);
    Oc[base + 16 + n] = f2bf(o1[r] * li);
    Oc[base + 32 + n] = f2bf(o2[r] * li);
    Oc[base + 48 + n] = f2bf(o3[r] * li);
  }
}

// ---------------------------------------------------------------------------
// Kernel 4: output projection  out = Oc · Wo^T  (f32 result)
//   One wave = 16x64 tile: one A-frag feeds 4 WMMAs; double-buffered K-loop.
// ---------------------------------------------------------------------------
__global__ void __launch_bounds__(32)
out_proj_kernel(const __bf16* __restrict__ Ocb, const __bf16* __restrict__ wo,
                float* __restrict__ out) {
  const int n0   = blockIdx.x * 64;
  const int r0   = blockIdx.y * 16;
  const int lane = threadIdx.x & 31;
  v8f c0 = {}, c1 = {}, c2 = {}, c3 = {};

  v16bf a  = load_a_frag(Ocb, D, r0, 0);
  v16bf b0 = load_b_frag(wo, D, n0 + 0,  0);
  v16bf b1 = load_b_frag(wo, D, n0 + 16, 0);
  v16bf b2 = load_b_frag(wo, D, n0 + 32, 0);
  v16bf b3 = load_b_frag(wo, D, n0 + 48, 0);

  for (int k0 = 0; k0 < D - 32; k0 += 32) {
    v16bf a2  = load_a_frag(Ocb, D, r0, k0 + 32);
    v16bf b02 = load_b_frag(wo, D, n0 + 0,  k0 + 32);
    v16bf b12 = load_b_frag(wo, D, n0 + 16, k0 + 32);
    v16bf b22 = load_b_frag(wo, D, n0 + 32, k0 + 32);
    v16bf b32 = load_b_frag(wo, D, n0 + 48, k0 + 32);
    c0 = wmma_bf16(a, b0, c0);
    c1 = wmma_bf16(a, b1, c1);
    c2 = wmma_bf16(a, b2, c2);
    c3 = wmma_bf16(a, b3, c3);
    a = a2; b0 = b02; b1 = b12; b2 = b22; b3 = b32;
  }
  c0 = wmma_bf16(a, b0, c0);
  c1 = wmma_bf16(a, b1, c1);
  c2 = wmma_bf16(a, b2, c2);
  c3 = wmma_bf16(a, b3, c3);

  const int n = lane & 15, mo = (lane >> 4) * 8;
#pragma unroll
  for (int r = 0; r < 8; ++r) {
    size_t row = (size_t)(r0 + mo + r) * D + n0 + n;
    out[row + 0]  = c0[r];
    out[row + 16] = c1[r];
    out[row + 32] = c2[r];
    out[row + 48] = c3[r];
  }
}

// ---------------------------------------------------------------------------
// Launch
// ---------------------------------------------------------------------------
extern "C" void kernel_launch(void* const* d_in, const int* in_sizes, int n_in,
                              void* d_out, int out_size, void* d_ws,
                              size_t ws_size, hipStream_t stream) {
  const float* x   = (const float*)d_in[0];
  const int*   pos = (const int*)  d_in[1];
  const float* Wq  = (const float*)d_in[2];
  const float* Wk  = (const float*)d_in[3];
  const float* Wv  = (const float*)d_in[4];
  const float* Wo  = (const float*)d_in[5];

  // workspace layout (bytes), all 256-aligned; total = 48 MiB
  constexpr size_t XB  = (size_t)B * S * D * 2;   // 8 MiB
  constexpr size_t WB  = (size_t)D * D * 2;       // 2 MiB
  char* ws = (char*)d_ws;
  __bf16* xb  = (__bf16*)(ws);
  __bf16* wqb = (__bf16*)(ws + XB);
  __bf16* wkb = (__bf16*)(ws + XB + WB);
  __bf16* wvb = (__bf16*)(ws + XB + 2 * WB);
  __bf16* wob = (__bf16*)(ws + XB + 3 * WB);
  __bf16* Qh  = (__bf16*)(ws + XB + 4 * WB);
  __bf16* Kh  = (__bf16*)(ws + 2 * XB + 4 * WB);
  __bf16* Vt  = (__bf16*)(ws + 3 * XB + 4 * WB);
  __bf16* Oc  = (__bf16*)(ws + 4 * XB + 4 * WB);
  if (ws_size < 5 * XB + 4 * WB) return;

  const int nx4 = (B * S * D) / 4;
  const int nw4 = (D * D) / 4;
  cvt_f32_bf16_kernel<<<(nx4 + 255) / 256, 256, 0, stream>>>(x,  xb,  nx4);
  cvt_f32_bf16_kernel<<<(nw4 + 255) / 256, 256, 0, stream>>>(Wq, wqb, nw4);
  cvt_f32_bf16_kernel<<<(nw4 + 255) / 256, 256, 0, stream>>>(Wk, wkb, nw4);
  cvt_f32_bf16_kernel<<<(nw4 + 255) / 256, 256, 0, stream>>>(Wv, wvb, nw4);
  cvt_f32_bf16_kernel<<<(nw4 + 255) / 256, 256, 0, stream>>>(Wo, wob, nw4);

  proj_qkv_kernel<<<dim3(D / 16, (B * S) / 16), 32, 0, stream>>>(
      xb, wqb, wkb, wvb, pos, Qh, Kh, Vt);

  flash_attn_kernel<<<dim3((B * H * S) / 16), 32, 0, stream>>>(Qh, Kh, Vt, Oc);

  out_proj_kernel<<<dim3(D / 16 / 4, (B * S) / 16), 32, 0, stream>>>(
      Oc, wob, (float*)d_out);
}
